// DiffWave_19061064860076
// MI455X (gfx1250) — matrix operand order repaired and checked by reading, not compiled, not run on hardware
//
#include <hip/hip_runtime.h>

#define C_CH    64
#define TWO_C   128
#define NLAY    30
#define T_LEN   16384
#define BATCH   4
#define TILE_N  64
#define BPAD    72      // ic-padded row for B/Z tiles (144B rows, 16B aligned)
#define YPAD    68

typedef __attribute__((ext_vector_type(16))) __bf16 v16bf;
typedef __attribute__((ext_vector_type(8)))  float  v8f;

union Frag16 {
    v16bf v;
    __bf16 h[16];
    uint4 q[2];
};

__device__ __forceinline__ float siluf(float x) {
    return x / (1.0f + __expf(-x));
}

// CDNA5 async global->LDS copy (ASYNCcnt-tracked), ISA §10.7 / §15.18.3
__device__ __forceinline__ void async_ld16(unsigned ldsOff, const void* g) {
    asm volatile("global_load_async_to_lds_b128 %0, %1, off"
                 :: "v"(ldsOff), "v"(g) : "memory");
}
__device__ __forceinline__ void wait_async0() {
    asm volatile("s_wait_asynccnt 0" ::: "memory");
}

// ---------------------------------------------------------------------------
// Diffusion-step embedding + FC stack + per-layer cond projections and the
// folded dilated-conv cond bias  ybias[l][b][m] = sum_tap,ic W*cond.
// ---------------------------------------------------------------------------
__global__ __launch_bounds__(256) void k_emb(const int* __restrict__ dstep,
                                             const float* __restrict__ p1W,
                                             const float* __restrict__ p1b,
                                             const float* __restrict__ p2W,
                                             const float* __restrict__ p2b,
                                             const float* __restrict__ dpW,
                                             const float* __restrict__ dpb,
                                             const float* __restrict__ dwW,
                                             float* __restrict__ cond,
                                             float* __restrict__ ybias) {
    __shared__ float emb[128];
    __shared__ float h1[512];
    __shared__ float h2[512];
    __shared__ float scond[NLAY * C_CH];
    const int tid = threadIdx.x;
    for (int b = 0; b < BATCH; ++b) {
        float step = (float)dstep[b];
        if (tid < 64) {
            float t = step * __expf((float)tid * (4.0f / 63.0f) * 2.302585093f);
            emb[tid]      = __sinf(t);
            emb[tid + 64] = __cosf(t);
        }
        __syncthreads();
        for (int o = tid; o < 512; o += 256) {
            float a = p1b[o];
            for (int j = 0; j < 128; ++j) a += p1W[o * 128 + j] * emb[j];
            h1[o] = siluf(a);
        }
        __syncthreads();
        for (int o = tid; o < 512; o += 256) {
            float a = p2b[o];
            for (int j = 0; j < 512; ++j) a += p2W[o * 512 + j] * h1[j];
            h2[o] = siluf(a);
        }
        __syncthreads();
        for (int idx = tid; idx < NLAY * C_CH; idx += 256) {
            int l = idx / C_CH, c = idx % C_CH;
            float a = dpb[l * C_CH + c];
            const float* w = dpW + ((size_t)(l * C_CH + c)) * 512;
            for (int j = 0; j < 512; ++j) a += w[j] * h2[j];
            scond[idx] = a;
            cond[((size_t)l * BATCH + b) * C_CH + c] = a;
        }
        __syncthreads();
        for (int idx = tid; idx < NLAY * TWO_C; idx += 256) {
            int l = idx / TWO_C, m = idx % TWO_C;
            const float* wl = dwW + ((size_t)(l * TWO_C + m)) * C_CH * 3;
            const float* cl = scond + l * C_CH;
            float a = 0.0f;
            for (int ic = 0; ic < C_CH; ++ic)
                a += (wl[ic * 3 + 0] + wl[ic * 3 + 1] + wl[ic * 3 + 2]) * cl[ic];
            ybias[((size_t)l * BATCH + b) * TWO_C + m] = a;
        }
        __syncthreads();
    }
}

// ---------------------------------------------------------------------------
// One-shot weight prep: dw_W / op_W -> bf16 WMMA A-fragment layout
// ---------------------------------------------------------------------------
__global__ __launch_bounds__(256) void k_wprep_dw(const float* __restrict__ dwW,
                                                  __bf16* __restrict__ dwF) {
    int idx = blockIdx.x * 256 + threadIdx.x;          // < 30*3*2*8*32*16
    int e    = idx & 15;
    int lane = (idx >> 4) & 31;
    int wave = (idx >> 9) & 7;
    int kk   = (idx >> 12) & 1;
    int tl   = idx >> 13;                              // tap + 3*layer
    int tap  = tl % 3;
    int layer = tl / 3;
    int hi   = lane >> 4;
    int row  = wave * 16 + (lane & 15);
    int K    = (hi << 4) + e;
    int ic   = kk * 32 + K;
    dwF[idx] = (__bf16)dwW[((size_t)(layer * TWO_C + row) * C_CH + ic) * 3 + tap];
}

__global__ __launch_bounds__(256) void k_wprep_op(const float* __restrict__ opW,
                                                  __bf16* __restrict__ opF) {
    int idx = blockIdx.x * 256 + threadIdx.x;          // < 30*2*8*32*16
    int e    = idx & 15;
    int lane = (idx >> 4) & 31;
    int wave = (idx >> 9) & 7;
    int kk   = (idx >> 12) & 1;
    int layer = idx >> 13;
    int hi   = lane >> 4;
    int row  = wave * 16 + (lane & 15);
    int K    = (hi << 4) + e;
    int ic   = kk * 32 + K;
    opF[idx] = (__bf16)opW[(size_t)(layer * TWO_C + row) * C_CH + ic];
}

// ---------------------------------------------------------------------------
// Input pointwise conv 1->64 + ReLU; writes f32 x and bf16 time-major mirror
// ---------------------------------------------------------------------------
__global__ __launch_bounds__(256) void k_in(const float* __restrict__ audio,
                                            const float* __restrict__ inW,
                                            const float* __restrict__ inb,
                                            float* __restrict__ x,
                                            __bf16* __restrict__ xbfT) {
    int idx = blockIdx.x * 256 + threadIdx.x;       // b*C*T + c*T + t
    int b = idx >> 20;
    int c = (idx >> 14) & 63;
    int t = idx & (T_LEN - 1);
    float v = audio[b * T_LEN + t] * inW[c] + inb[c];
    v = v > 0.0f ? v : 0.0f;
    x[idx] = v;
    xbfT[((size_t)b * T_LEN + t) * C_CH + c] = (__bf16)v;
}

// ---------------------------------------------------------------------------
// Fused residual layer.  One block = one batch x 64-timestep tile, 8 waves.
// Interior tiles: DMA bf16 x into LDS (async-to-LDS), cond folded into ybias.
// Boundary tiles: f32 load + cond + cvt (zero-padding correct).
// ---------------------------------------------------------------------------
__global__ __launch_bounds__(256) void k_layer(const float* __restrict__ xIn,
                                               const __bf16* __restrict__ xbfIn,
                                               float* __restrict__ xOut,
                                               __bf16* __restrict__ xbfOut,
                                               float* __restrict__ skip,
                                               const float* __restrict__ cond,
                                               const float* __restrict__ ybias,
                                               const __bf16* __restrict__ dwF,
                                               const float* __restrict__ dwb,
                                               const __bf16* __restrict__ opF,
                                               const float* __restrict__ opb,
                                               int layer, int dil, int first) {
    __shared__ __bf16 sB[3 * TILE_N * BPAD];   // [tap][n][icPad]
    __shared__ float  sY[TWO_C * YPAD];        // [row][n]
    __bf16* sZ = sB;                           // [n][icPad] (overlay)

    const int tid  = threadIdx.x;
    const int lane = tid & 31;
    const int wave = tid >> 5;
    const int hi   = lane >> 4;
    const int ln   = lane & 15;
    const int mBase = wave * 16;

    const int b  = blockIdx.y;
    const int t0 = blockIdx.x * TILE_N;
    const bool fast = (t0 >= dil) && (t0 + TILE_N + dil <= T_LEN);

    const float* xb = xIn + (size_t)b * C_CH * T_LEN;

    // ---- A fragments: coalesced vector loads from prepped bf16 buffers ----
    Frag16 aF[3][2];
    Frag16 oF[2];
#pragma unroll
    for (int tap = 0; tap < 3; ++tap)
#pragma unroll
        for (int kk = 0; kk < 2; ++kk) {
            const uint4* p = (const uint4*)(dwF +
                ((size_t)((((layer * 3 + tap) * 2 + kk) * 8 + wave) * 32 + lane)) * 16);
            aF[tap][kk].q[0] = p[0];
            aF[tap][kk].q[1] = p[1];
        }
#pragma unroll
    for (int kk = 0; kk < 2; ++kk) {
        const uint4* p = (const uint4*)(opF +
            ((size_t)(((layer * 2 + kk) * 8 + wave) * 32 + lane)) * 16);
        oF[kk].q[0] = p[0];
        oF[kk].q[1] = p[1];
    }

    // ---- stage B tiles ----------------------------------------------------
    if (fast) {
        // pure DMA: 3 taps x 64 rows x 128B from bf16 time-major mirror
        const __bf16* xt = xbfIn + (size_t)b * T_LEN * C_CH;
        for (int idx = tid; idx < 3 * TILE_N * 8; idx += 256) {
            int chunk = idx & 7;               // 8 x 16B per 64-ch row
            int row   = idx >> 3;              // tap*64 + n
            int tap   = row >> 6;
            int n     = row & 63;
            int t = t0 + n + (tap - 1) * dil;
            const void* g = xt + (size_t)t * C_CH + chunk * 8;
            unsigned lds = (unsigned)(uintptr_t)&sB[row * BPAD] + chunk * 16;
            async_ld16(lds, g);
        }
        wait_async0();
    } else {
        const float* condp = cond + ((size_t)layer * BATCH + b) * C_CH;
        for (int idx = tid; idx < 3 * C_CH * TILE_N; idx += 256) {
            int n   = idx & (TILE_N - 1);
            int r   = idx >> 6;
            int tap = r >> 6;
            int ic  = r & 63;
            int t = t0 + n + (tap - 1) * dil;
            float v = 0.0f;
            if (t >= 0 && t < T_LEN) v = xb[ic * T_LEN + t] + condp[ic];
            sB[(tap * TILE_N + n) * BPAD + ic] = (__bf16)v;
        }
    }
    __syncthreads();

    // ---- dilated-conv GEMM: y[128 x 64] = sum_tap W_tap @ B_tap -----------
    const v8f vzero = {0.f, 0.f, 0.f, 0.f, 0.f, 0.f, 0.f, 0.f};
    v8f acc[4];
#pragma unroll
    for (int nt = 0; nt < 4; ++nt) acc[nt] = vzero;

#pragma unroll
    for (int kk = 0; kk < 2; ++kk)
#pragma unroll
        for (int tap = 0; tap < 3; ++tap) {
            v16bf a = aF[tap][kk].v;
#pragma unroll
            for (int nt = 0; nt < 4; ++nt) {
                const __bf16* bp =
                    &sB[(tap * TILE_N + nt * 16 + ln) * BPAD + kk * 32 + hi * 16];
                Frag16 bF;
                bF.q[0] = *(const uint4*)(bp);
                bF.q[1] = *(const uint4*)(bp + 8);
                acc[nt] = __builtin_amdgcn_wmma_f32_16x16x32_bf16(
                    false, a, false, bF.v, (short)0, acc[nt], false, false);
            }
        }

    // ---- y (+bias, +folded cond bias on fast path) to LDS -----------------
    const float* yb = ybias + ((size_t)layer * BATCH + b) * TWO_C;
#pragma unroll
    for (int nt = 0; nt < 4; ++nt) {
        int col = nt * 16 + ln;
#pragma unroll
        for (int r = 0; r < 8; ++r) {
            int row = mBase + r + (hi << 3);
            float bias = dwb[layer * TWO_C + row] + (fast ? yb[row] : 0.0f);
            sY[row * YPAD + col] = acc[nt][r] + bias;
        }
    }
    __syncthreads();

    // ---- gate: z = sigmoid(y[0:64]) * tanh(y[64:128]) ---------------------
    for (int idx = tid; idx < C_CH * TILE_N; idx += 256) {
        int c = idx >> 6, n = idx & 63;
        float g = sY[c * YPAD + n];
        float f = sY[(c + C_CH) * YPAD + n];
        float sig = __builtin_amdgcn_rcpf(1.0f + __expf(-g));
        float e2f = __expf(2.0f * f);
        float th  = 1.0f - 2.0f * __builtin_amdgcn_rcpf(e2f + 1.0f);
        sZ[n * BPAD + c] = (__bf16)(sig * th);
    }
    __syncthreads();

    // ---- pointwise op GEMM: o[128 x 64] = opW @ z --------------------------
    v8f acc2[4];
#pragma unroll
    for (int nt = 0; nt < 4; ++nt) acc2[nt] = vzero;
#pragma unroll
    for (int kk = 0; kk < 2; ++kk) {
        v16bf a = oF[kk].v;
#pragma unroll
        for (int nt = 0; nt < 4; ++nt) {
            const __bf16* bp = &sZ[(nt * 16 + ln) * BPAD + kk * 32 + hi * 16];
            Frag16 bF;
            bF.q[0] = *(const uint4*)(bp);
            bF.q[1] = *(const uint4*)(bp + 8);
            acc2[nt] = __builtin_amdgcn_wmma_f32_16x16x32_bf16(
                false, a, false, bF.v, (short)0, acc2[nt], false, false);
        }
    }

    // ---- epilogue: waves 0-3 residual (+bf16 mirror), waves 4-7 skip ------
#pragma unroll
    for (int nt = 0; nt < 4; ++nt) {
        int col = nt * 16 + ln;
        int t = t0 + col;
        if (mBase < C_CH) {
            __bf16 pk[8];
#pragma unroll
            for (int r = 0; r < 8; ++r) {
                int row = mBase + r + (hi << 3);
                float v = acc2[nt][r] + opb[layer * TWO_C + row];
                size_t xi = ((size_t)b * C_CH + row) * T_LEN + t;
                float xn = (xIn[xi] + v) * 0.70710678118654752f;
                xOut[xi] = xn;
                pk[r] = (__bf16)xn;
            }
            // 8 consecutive channels at fixed t -> one 16B store
            *(uint4*)&xbfOut[((size_t)b * T_LEN + t) * C_CH + mBase + hi * 8] =
                *(const uint4*)pk;
        } else {
#pragma unroll
            for (int r = 0; r < 8; ++r) {
                int row = mBase + r + (hi << 3);
                float v = acc2[nt][r] + opb[layer * TWO_C + row];
                size_t si = ((size_t)b * C_CH + (row - C_CH)) * T_LEN + t;
                skip[si] = first ? v : (skip[si] + v);
            }
        }
    }
}

// ---------------------------------------------------------------------------
// Output head: relu(skW @ (skip/sqrt(30)) + skb) -> outW dot + outb
// ---------------------------------------------------------------------------
__global__ __launch_bounds__(256) void k_out(const float* __restrict__ skip,
                                             const float* __restrict__ skW,
                                             const float* __restrict__ skb,
                                             const float* __restrict__ outW,
                                             const float* __restrict__ outb,
                                             float* __restrict__ out) {
    __shared__ float sW[C_CH * C_CH];
    __shared__ float sBias[C_CH];
    __shared__ float sOW[C_CH];
    const int tid = threadIdx.x;
    for (int i = tid; i < C_CH * C_CH; i += 256) sW[i] = skW[i];
    if (tid < C_CH) { sBias[tid] = skb[tid]; sOW[tid] = outW[tid]; }
    __syncthreads();

    int idx = blockIdx.x * 256 + tid;               // b*T + t
    int b = idx >> 14;
    int t = idx & (T_LEN - 1);
    const float inv = 0.18257418583505537f;         // 1/sqrt(30)

    float s[C_CH];
#pragma unroll
    for (int ic = 0; ic < C_CH; ++ic)
        s[ic] = skip[((size_t)b * C_CH + ic) * T_LEN + t] * inv;

    float o = outb[0];
    for (int c = 0; c < C_CH; ++c) {
        float a = sBias[c];
#pragma unroll
        for (int ic = 0; ic < C_CH; ++ic) a += sW[c * C_CH + ic] * s[ic];
        a = a > 0.0f ? a : 0.0f;
        o += sOW[c] * a;
    }
    out[idx] = o;
}

// ---------------------------------------------------------------------------
extern "C" void kernel_launch(void* const* d_in, const int* in_sizes, int n_in,
                              void* d_out, int out_size, void* d_ws, size_t ws_size,
                              hipStream_t stream) {
    (void)in_sizes; (void)n_in; (void)out_size; (void)ws_size;
    const float* audio = (const float*)d_in[0];
    const int*   dstep = (const int*)d_in[1];
    const float* inW   = (const float*)d_in[2];
    const float* inb   = (const float*)d_in[3];
    const float* p1W   = (const float*)d_in[4];
    const float* p1b   = (const float*)d_in[5];
    const float* p2W   = (const float*)d_in[6];
    const float* p2b   = (const float*)d_in[7];
    const float* dwW   = (const float*)d_in[8];
    const float* dwb   = (const float*)d_in[9];
    const float* dpW   = (const float*)d_in[10];
    const float* dpb   = (const float*)d_in[11];
    const float* opW   = (const float*)d_in[12];
    const float* opb   = (const float*)d_in[13];
    const float* skW   = (const float*)d_in[14];
    const float* skb   = (const float*)d_in[15];
    const float* outW  = (const float*)d_in[16];
    const float* outb  = (const float*)d_in[17];
    float* out = (float*)d_out;

    const size_t TEN = (size_t)BATCH * C_CH * T_LEN;   // 4.19M elems
    float* ws    = (float*)d_ws;
    float* xA    = ws;
    float* xB    = xA + TEN;
    float* skip  = xB + TEN;
    float* cond  = skip + TEN;                         // 7680 floats
    float* ybias = cond + (size_t)NLAY * BATCH * C_CH; // 15360 floats
    __bf16* dwF  = (__bf16*)(ybias + (size_t)NLAY * BATCH * TWO_C);
    __bf16* opF  = dwF + (size_t)NLAY * 3 * 2 * 8 * 32 * 16;   // 737280
    __bf16* xbfA = opF + (size_t)NLAY * 2 * 8 * 32 * 16;       // 245760
    __bf16* xbfB = xbfA + TEN;

    const int NDWF = NLAY * 3 * 2 * 8 * 32 * 16;       // 737280
    const int NOPF = NLAY * 2 * 8 * 32 * 16;           // 245760

    k_wprep_dw<<<NDWF / 256, 256, 0, stream>>>(dwW, dwF);
    k_wprep_op<<<NOPF / 256, 256, 0, stream>>>(opW, opF);
    k_emb<<<1, 256, 0, stream>>>(dstep, p1W, p1b, p2W, p2b, dpW, dpb, dwW,
                                 cond, ybias);
    k_in<<<(int)(TEN / 256), 256, 0, stream>>>(audio, inW, inb, xA, xbfA);

    float* xi = xA;    float* xo = xB;
    __bf16* hi = xbfA; __bf16* ho = xbfB;
    for (int l = 0; l < NLAY; ++l) {
        int d = 1 << (l % 10);
        dim3 grid(T_LEN / TILE_N, BATCH);
        k_layer<<<grid, 256, 0, stream>>>(xi, hi, xo, ho, skip, cond, ybias,
                                          dwF, dwb, opF, opb, l, d, l == 0 ? 1 : 0);
        float* tf = xi; xi = xo; xo = tf;
        __bf16* tb = hi; hi = ho; ho = tb;
    }

    k_out<<<(BATCH * T_LEN) / 256, 256, 0, stream>>>(skip, skW, skb, outW, outb, out);
}
